// Seq2SeqAttentionDecoder_76467597738452
// MI455X (gfx1250) — matrix operand order repaired and checked by reading, not compiled, not run on hardware
//
#include <hip/hip_runtime.h>
#include <hip/hip_bf16.h>

// ---------------------------------------------------------------------------
// Seq2Seq attention decoder (additive attention + 2-layer GRU + big vocab
// projection) for MI455X / gfx1250.
//
// Heavy GEMMs (k_proj, logits) use v_wmma_f32_16x16x32_bf16 (fp32 accum).
// Logits kernel: 16x64 output strip per wave (4 accumulators reuse one A
// fragment per k-step) -> 4x less A traffic; output-store-bound overall.
// Sequential recurrence stays fp32 VALU (latency-bound, tiny).
// ---------------------------------------------------------------------------

typedef __attribute__((ext_vector_type(16))) __bf16 v16bf;
typedef __attribute__((ext_vector_type(8)))  float  v8f;

#define Bc 64
#define Tc 32
#define Sc 128
#define Hc 256
#define Ec 256
#define Vc 32000
#define NEGC (-1e6f)

// round-to-nearest-even f32 -> bf16 without relying on __bf16 cast support
__device__ inline __bf16 f32_to_bf16(float f) {
  unsigned u = __builtin_bit_cast(unsigned, f);
  unsigned r = u + 0x7FFFu + ((u >> 16) & 1u);
  unsigned short h = (unsigned short)(r >> 16);
  return __builtin_bit_cast(__bf16, h);
}

__global__ void cvt_bf16_kernel(const float* __restrict__ s, __bf16* __restrict__ d, int n) {
  int i = blockIdx.x * blockDim.x + threadIdx.x;
  if (i < n) d[i] = f32_to_bf16(s[i]);
}

__global__ void init_hidden_kernel(const float* __restrict__ hs,
                                   float* __restrict__ h0, float* __restrict__ h1) {
  int i = blockIdx.x * blockDim.x + threadIdx.x;
  if (i < Bc * Hc) { h0[i] = hs[i]; h1[i] = hs[Bc * Hc + i]; }
}

// --------------------------- WMMA fragment loads ---------------------------
// A (16x32 bf16, M x K): lane l -> row m=l&15, sel=l>>4.
//   elems 0..7  = A[m][k0 + 8*sel + 0..7]
//   elems 8..15 = A[m][k0 + 16 + 8*sel + 0..7]
__device__ inline v16bf load_a_frag(const __bf16* __restrict__ A, int lda,
                                    int row, int k0, int sel) {
  const __bf16* p = A + (size_t)row * lda + k0 + sel * 8;
  union { v16bf v; uint4 q[2]; } u;
  u.q[0] = *reinterpret_cast<const uint4*>(p);
  u.q[1] = *reinterpret_cast<const uint4*>(p + 16);
  return u.v;
}

// B (32x16 bf16, K x N) with B = W^T, W [N,K] row-major: lane l -> col n=l&15.
//   elems 0..7 = W[n][k0 + 8*sel + 0..7], elems 8..15 = W[n][k0 + 16 + 8*sel + 0..7]
__device__ inline v16bf load_b_frag(const __bf16* __restrict__ W, int ldw,
                                    int col, int k0, int sel) {
  const __bf16* p = W + (size_t)col * ldw + k0 + sel * 8;
  union { v16bf v; uint4 q[2]; } u;
  u.q[0] = *reinterpret_cast<const uint4*>(p);
  u.q[1] = *reinterpret_cast<const uint4*>(p + 16);
  return u.v;
}

// k_proj = enc @ Wk^T   (A [M,K] bf16 row-major, W [N,K] bf16 row-major, C f32)
__global__ void wmma_kproj_kernel(const __bf16* __restrict__ A, const __bf16* __restrict__ W,
                                  float* __restrict__ C, int M, int N, int K) {
  int ntn = N >> 4;
  int tm = blockIdx.x / ntn, tn = blockIdx.x % ntn;
  int lane = threadIdx.x;
  int nl = lane & 15, sel = lane >> 4;
  v8f acc = {};
  for (int k0 = 0; k0 < K; k0 += 32) {
    v16bf a = load_a_frag(A, K, tm * 16 + nl, k0, sel);
    v16bf b = load_b_frag(W, K, tn * 16 + nl, k0, sel);
    acc = __builtin_amdgcn_wmma_f32_16x16x32_bf16(false, a, false, b,
                                                  (short)0, acc, false, false);
  }
  int col = tn * 16 + nl;
#pragma unroll
  for (int r = 0; r < 8; ++r) {
    int m = tm * 16 + sel * 8 + r;
    C[(size_t)m * N + col] = acc[r];
  }
}

// logits = outs @ Wd^T + bd ; rows ordered (t,b) -> store [B,T,V]
// One wave computes a 16x64 output strip: 4 WMMA accumulators share each
// A fragment per k-step (A is the hot operand; 4x fewer A loads).
__global__ void wmma_logits_kernel(const __bf16* __restrict__ A, const __bf16* __restrict__ Wd,
                                   const float* __restrict__ bd, float* __restrict__ out) {
  const int K = Hc;
  const int nstrips = Vc / 64;  // 500
  int tm = blockIdx.x / nstrips, ts = blockIdx.x % nstrips;
  int lane = threadIdx.x;
  int nl = lane & 15, sel = lane >> 4;
  int arow = tm * 16 + nl;
  int c0 = ts * 64 + nl;
  v8f acc0 = {}, acc1 = {}, acc2 = {}, acc3 = {};
#pragma unroll
  for (int k0 = 0; k0 < K; k0 += 32) {
    v16bf a  = load_a_frag(A, K, arow, k0, sel);
    v16bf b0 = load_b_frag(Wd, K, c0,      k0, sel);
    v16bf b1 = load_b_frag(Wd, K, c0 + 16, k0, sel);
    v16bf b2 = load_b_frag(Wd, K, c0 + 32, k0, sel);
    v16bf b3 = load_b_frag(Wd, K, c0 + 48, k0, sel);
    acc0 = __builtin_amdgcn_wmma_f32_16x16x32_bf16(false, a, false, b0, (short)0, acc0, false, false);
    acc1 = __builtin_amdgcn_wmma_f32_16x16x32_bf16(false, a, false, b1, (short)0, acc1, false, false);
    acc2 = __builtin_amdgcn_wmma_f32_16x16x32_bf16(false, a, false, b2, (short)0, acc2, false, false);
    acc3 = __builtin_amdgcn_wmma_f32_16x16x32_bf16(false, a, false, b3, (short)0, acc3, false, false);
  }
  float bv0 = bd[c0], bv1 = bd[c0 + 16], bv2 = bd[c0 + 32], bv3 = bd[c0 + 48];
#pragma unroll
  for (int r = 0; r < 8; ++r) {
    int m = tm * 16 + sel * 8 + r;   // row index in [T*B]
    int t = m / Bc, b = m % Bc;
    float* op = out + ((size_t)b * Tc + t) * Vc;
    op[c0]      = acc0[r] + bv0;
    op[c0 + 16] = acc1[r] + bv1;
    op[c0 + 32] = acc2[r] + bv2;
    op[c0 + 48] = acc3[r] + bv3;
  }
}

// ------------------- per-step attention + input assembly -------------------
// one block per batch element; 256 threads
__global__ void attn_step_kernel(const float* __restrict__ h1, const float* __restrict__ kproj,
                                 const float* __restrict__ enc, const float* __restrict__ Wq,
                                 const float* __restrict__ wv, const int* __restrict__ X,
                                 const float* __restrict__ emb, const int* __restrict__ vlen,
                                 float* __restrict__ inp, int t) {
  int b = blockIdx.x, tid = threadIdx.x;
  __shared__ float q[Hc];
  __shared__ float sc[Sc];
  __shared__ float smax, ssum;
  // q_proj = h1[b] @ Wq^T
  {
    const float* hb = h1 + b * Hc;
    const float* wr = Wq + (size_t)tid * Hc;
    float acc = 0.f;
    for (int k = 0; k < Hc; ++k) acc += hb[k] * wr[k];
    q[tid] = acc;
  }
  __syncthreads();
  // scores[s] = wv . tanh(q + kproj[b,s])  (masked)
  if (tid < Sc) {
    const float* kp = kproj + ((size_t)b * Sc + tid) * Hc;
    float s = 0.f;
    for (int h = 0; h < Hc; ++h) s += tanhf(q[h] + kp[h]) * wv[h];
    sc[tid] = (tid < vlen[b]) ? s : NEGC;
  }
  __syncthreads();
  if (tid == 0) {
    float m = sc[0];
    for (int s = 1; s < Sc; ++s) m = fmaxf(m, sc[s]);
    float su = 0.f;
    for (int s = 0; s < Sc; ++s) su += expf(sc[s] - m);
    smax = m; ssum = su;
  }
  __syncthreads();
  if (tid < Sc) sc[tid] = expf(sc[tid] - smax) / ssum;
  __syncthreads();
  // context[h] = sum_s attn[s] * enc[b,s,h] ; inp = [context, emb[X[b,t]]]
  float ctx = 0.f;
  for (int s = 0; s < Sc; ++s) ctx += sc[s] * enc[((size_t)b * Sc + s) * Hc + tid];
  float* ib = inp + (size_t)b * (Hc + Ec);
  ib[tid]      = ctx;
  ib[Hc + tid] = emb[(size_t)X[b * Tc + t] * Ec + tid];
}

// ------------------------------- GRU cell ----------------------------------
// one block per batch element; thread j computes output feature j
__global__ void gru_cell_kernel(const float* __restrict__ x, int Kin,
                                const float* __restrict__ h,
                                const float* __restrict__ Wih, const float* __restrict__ Whh,
                                const float* __restrict__ bih, const float* __restrict__ bhh,
                                float* __restrict__ hnew,
                                __bf16* __restrict__ outs_b, int t) {
  int b = blockIdx.x, j = threadIdx.x;
  __shared__ float xs[Hc + Ec];
  __shared__ float hs[Hc];
  for (int k = j; k < Kin; k += blockDim.x) xs[k] = x[(size_t)b * Kin + k];
  hs[j] = h[b * Hc + j];
  __syncthreads();
  float ir = bih[j], iz = bih[Hc + j], in_ = bih[2 * Hc + j];
  const float* w0 = Wih + (size_t)j * Kin;
  const float* w1 = Wih + (size_t)(Hc + j) * Kin;
  const float* w2 = Wih + (size_t)(2 * Hc + j) * Kin;
  for (int k = 0; k < Kin; ++k) {
    float xv = xs[k];
    ir += xv * w0[k]; iz += xv * w1[k]; in_ += xv * w2[k];
  }
  float hr = bhh[j], hz = bhh[Hc + j], hn = bhh[2 * Hc + j];
  const float* u0 = Whh + (size_t)j * Hc;
  const float* u1 = Whh + (size_t)(Hc + j) * Hc;
  const float* u2 = Whh + (size_t)(2 * Hc + j) * Hc;
  for (int k = 0; k < Hc; ++k) {
    float hv = hs[k];
    hr += hv * u0[k]; hz += hv * u1[k]; hn += hv * u2[k];
  }
  float r = 1.f / (1.f + expf(-(ir + hr)));
  float z = 1.f / (1.f + expf(-(iz + hz)));
  float n = tanhf(in_ + r * hn);
  float o = (1.f - z) * n + z * hs[j];
  hnew[b * Hc + j] = o;
  if (outs_b) {
    outs_b[((size_t)t * Bc + b) * Hc + j] = f32_to_bf16(o);
  }
}

__global__ void hfinal_kernel(const float* __restrict__ h0, const float* __restrict__ h1,
                              float* __restrict__ out) {
  int i = blockIdx.x * blockDim.x + threadIdx.x;
  if (i < Bc * Hc) { out[i] = h0[i]; out[Bc * Hc + i] = h1[i]; }
}

// ---------------------------------------------------------------------------
extern "C" void kernel_launch(void* const* d_in, const int* in_sizes, int n_in,
                              void* d_out, int out_size, void* d_ws, size_t ws_size,
                              hipStream_t stream) {
  (void)in_sizes; (void)n_in; (void)out_size; (void)ws_size;
  const int*   X    = (const int*)d_in[0];
  const float* enc  = (const float*)d_in[1];
  const float* hs   = (const float*)d_in[2];
  const int*   vlen = (const int*)d_in[3];
  const float* emb  = (const float*)d_in[4];
  const float* Wq   = (const float*)d_in[5];
  const float* Wk   = (const float*)d_in[6];
  const float* wv   = (const float*)d_in[7];
  const float* Wih0 = (const float*)d_in[8];
  const float* Whh0 = (const float*)d_in[9];
  const float* bih0 = (const float*)d_in[10];
  const float* bhh0 = (const float*)d_in[11];
  const float* Wih1 = (const float*)d_in[12];
  const float* Whh1 = (const float*)d_in[13];
  const float* bih1 = (const float*)d_in[14];
  const float* bhh1 = (const float*)d_in[15];
  const float* Wd   = (const float*)d_in[16];
  const float* bd   = (const float*)d_in[17];
  float* out = (float*)d_out;

  char* base = (char*)d_ws;
  size_t off = 0;
  auto alloc = [&](size_t bytes) {
    void* p = base + off;
    off += (bytes + 255) & ~(size_t)255;
    return p;
  };
  __bf16* enc_bf  = (__bf16*)alloc((size_t)Bc * Sc * Hc * 2);
  __bf16* Wk_bf   = (__bf16*)alloc((size_t)Hc * Hc * 2);
  __bf16* Wd_bf   = (__bf16*)alloc((size_t)Vc * Hc * 2);
  float*  kproj   = (float*)alloc((size_t)Bc * Sc * Hc * 4);
  float*  inp     = (float*)alloc((size_t)Bc * (Hc + Ec) * 4);
  float*  h0a     = (float*)alloc((size_t)Bc * Hc * 4);
  float*  h0b     = (float*)alloc((size_t)Bc * Hc * 4);
  float*  h1a     = (float*)alloc((size_t)Bc * Hc * 4);
  float*  h1b     = (float*)alloc((size_t)Bc * Hc * 4);
  __bf16* outs_bf = (__bf16*)alloc((size_t)Tc * Bc * Hc * 2);

  int n;
  n = Bc * Sc * Hc; cvt_bf16_kernel<<<(n + 255) / 256, 256, 0, stream>>>(enc, enc_bf, n);
  n = Hc * Hc;      cvt_bf16_kernel<<<(n + 255) / 256, 256, 0, stream>>>(Wk, Wk_bf, n);
  n = Vc * Hc;      cvt_bf16_kernel<<<(n + 255) / 256, 256, 0, stream>>>(Wd, Wd_bf, n);
  init_hidden_kernel<<<(Bc * Hc + 255) / 256, 256, 0, stream>>>(hs, h0a, h1a);

  // k_proj = enc @ Wk^T : [B*S, H]
  {
    int M = Bc * Sc, N = Hc;
    wmma_kproj_kernel<<<(M / 16) * (N / 16), 32, 0, stream>>>(enc_bf, Wk_bf, kproj, M, N, Hc);
  }

  // sequential decode
  for (int t = 0; t < Tc; ++t) {
    float* h0c = (t & 1) ? h0b : h0a;
    float* h0n = (t & 1) ? h0a : h0b;
    float* h1c = (t & 1) ? h1b : h1a;
    float* h1n = (t & 1) ? h1a : h1b;
    attn_step_kernel<<<Bc, Hc, 0, stream>>>(h1c, kproj, enc, Wq, wv, X, emb, vlen, inp, t);
    gru_cell_kernel<<<Bc, Hc, 0, stream>>>(inp, Hc + Ec, h0c, Wih0, Whh0, bih0, bhh0,
                                           h0n, nullptr, t);
    gru_cell_kernel<<<Bc, Hc, 0, stream>>>(h0n, Hc, h1c, Wih1, Whh1, bih1, bhh1,
                                           h1n, outs_bf, t);
  }

  // logits (dominant GEMM; 262MB output store bound): 16x64 strip per wave
  wmma_logits_kernel<<<((Tc * Bc) / 16) * (Vc / 64), 32, 0, stream>>>(outs_bf, Wd_bf, bd, out);

  // T=32 even -> final hidden lives in the 'a' buffers
  hfinal_kernel<<<(Bc * Hc + 255) / 256, 256, 0, stream>>>(h0a, h1a, out + (size_t)Bc * Tc * Vc);
}
